// blobDice_and_CE_loss_9852654977450
// MI455X (gfx1250) — matrix operand order repaired and checked by reading, not compiled, not run on hardware
//
#include <hip/hip_runtime.h>
#include <hip/hip_bf16.h>

// ---------------------------------------------------------------------------
// blob-dice + CE loss, collapsed to per-(batch,category) sufficient stats.
// The category contraction  P(3 x N) * Indicator(N x 8)  is computed with
// v_wmma_f32_16x16x32_f16 (A rows: p0,p1,p2,ones ; B: category one-hot).
// ---------------------------------------------------------------------------

typedef __attribute__((ext_vector_type(16))) _Float16 v16h;
typedef __attribute__((ext_vector_type(8)))  float    v8f;

#define BATCH            2
#define NCH              3
#define NSP              (96*96*96)          /* 884736 spatial voxels per b */
#define KBLOBS           6
#define NCAT             7                   /* ml values 0..6              */
#define CHUNKS_PER_B     (NSP/32)            /* 27648                       */
#define WAVES_PER_B      1728
#define CHUNKS_PER_WAVE  (CHUNKS_PER_B/WAVES_PER_B)   /* 16                 */
#define NBLOCKS          ((WAVES_PER_B*BATCH)/8)      /* 432 blocks x 256t  */
#define WS_CE            56                  /* ws float index of CE sum    */

__global__ void zero_ws_kernel(float* __restrict__ ws) {
    if (threadIdx.x < 64) ws[threadIdx.x] = 0.0f;
}

__global__ __launch_bounds__(256)
void blob_accum_kernel(const float* __restrict__ x,
                       const int*   __restrict__ ml,
                       float*       __restrict__ ws)
{
    // per-wave staging: 32 voxels x {p0,p1,p2} f16 + cat u16
    __shared__ _Float16       sP[8][32][3];
    __shared__ unsigned short sC[8][32];

    const int warp  = threadIdx.x >> 5;
    const int lane  = threadIdx.x & 31;
    const int gw    = blockIdx.x * 8 + warp;
    const int b     = gw / WAVES_PER_B;            // batch element owned by wave
    const int wslot = gw - b * WAVES_PER_B;
    const int chunk0 = wslot * CHUNKS_PER_WAVE;    // 16 contiguous chunks

    const float* xb = x  + (size_t)b * NCH * NSP;
    const int*   mb = ml + (size_t)b * NSP;

    v8f   acc = {};     // f32 accumulator: D[c][n] = sum p_c over cat n; D[3][n]=cnt
    float ce  = 0.0f;

    const int m = lane & 15;    // A row (== B column n for this lane)
    const int h = lane >> 4;    // lane half selects K sub-range per ISA layout

    for (int i = 0; i < CHUNKS_PER_WAVE; ++i) {
        const int s = (chunk0 + i) * 32 + lane;    // spatial index, fully coalesced

        float x0 = xb[s];
        float x1 = xb[s + NSP];
        float x2 = xb[s + 2 * NSP];
        int   cat = mb[s];

        if (i + 1 < CHUNKS_PER_WAVE) {             // global_prefetch_b8 next chunk
            __builtin_prefetch(&xb[s + 32], 0, 1);
            __builtin_prefetch(&xb[s + 32 + NSP], 0, 1);
            __builtin_prefetch(&xb[s + 32 + 2 * NSP], 0, 1);
            __builtin_prefetch(&mb[s + 32], 0, 1);
        }

        float mx  = fmaxf(x0, fmaxf(x1, x2));
        float e0  = __expf(x0 - mx);
        float e1  = __expf(x1 - mx);
        float e2  = __expf(x2 - mx);
        float sum = e0 + e1 + e2;
        // Raw v_rcp_f32: ~1 ulp error, far below the f16 truncation applied
        // next; avoids the 12-op correctly-rounded divide sequence.
        float inv = __builtin_amdgcn_rcpf(sum);

        // cross-entropy term: -(x[y] - mx - log(sum)), y = (ml>0)
        float xy = (cat > 0) ? x1 : x0;
        ce += (mx + __logf(sum)) - xy;

        sP[warp][lane][0] = (_Float16)(e0 * inv);
        sP[warp][lane][1] = (_Float16)(e1 * inv);
        sP[warp][lane][2] = (_Float16)(e2 * inv);
        sC[warp][lane]    = (unsigned short)cat;
        __syncthreads();   // uniform trip count across all 8 waves

        // Build WMMA fragments per CDNA5 16-bit layouts.
        // A (16x32, row m): element e -> K = e<8 ? e+8h : e+8+8h
        // B (32x16, col n=m): element e -> K = e<8 ? e+8h : e+8+8h
        v16h Af, Bf;
#pragma unroll
        for (int e = 0; e < 16; ++e) {
            const int Ka = (e < 8) ? (e + 8 * h) : (e + 8 + 8 * h);
            _Float16 av = (_Float16)0.0f;
            if (m < 3)       av = sP[warp][Ka][m];   // p_m at voxel Ka
            else if (m == 3) av = (_Float16)1.0f;    // ones row -> counts
            Af[e] = av;
            Bf[e] = (sC[warp][Ka] == (unsigned short)m)
                        ? (_Float16)1.0f : (_Float16)0.0f;
        }

        acc = __builtin_amdgcn_wmma_f32_16x16x32_f16(
                  /*neg_a=*/false, Af, /*neg_b=*/false, Bf,
                  /*c_mod=*/(short)0, acc, /*reuse_a=*/false, /*reuse_b=*/false);
        __syncthreads();   // protect staging buffers before next iteration
    }

    // D layout: lane n (<16), VGPR r => D[r][n].  Rows 0..2 = channel sums,
    // row 3 = counts; rows 4..15 are zero (A rows zero).
    if (lane < NCAT) {
        float* dst = ws + ((b * NCAT) + lane) * 4;
        atomicAdd(dst + 0, acc[0]);
        atomicAdd(dst + 1, acc[1]);
        atomicAdd(dst + 2, acc[2]);
        atomicAdd(dst + 3, acc[3]);
    }

    // wave32 reduction of CE partial
#pragma unroll
    for (int off = 16; off; off >>= 1) ce += __shfl_xor(ce, off, 32);
    if (lane == 0) atomicAdd(ws + WS_CE, ce);
}

__global__ void finalize_kernel(const float* __restrict__ ws,
                                float*       __restrict__ out)
{
    if (threadIdx.x != 0) return;

    float blob = 0.0f;
    for (int b = 0; b < BATCH; ++b) {
        const float* wb = ws + b * NCAT * 4;
        const float S00 = wb[0], S01 = wb[1];           // sums over ml==0
        const float cnt0 = wb[3];
        for (int u = 1; u <= KBLOBS; ++u) {
            const float* t   = wb + u * 4;              // sums over ml==u
            const float cntu = t[3];
            const float nOther = (float)NSP - cnt0 - cntu;  // masked voxels
            const float third  = nOther * (1.0f / 3.0f);    // softmax(0)=1/3
            const float inter0 = S00 + third;
            const float inter1 = t[1];
            const float sp0    = S00 + t[0] + third;
            const float sp1    = S01 + t[1] + third;
            const float dc0 = 2.0f * inter0 / fmaxf(((float)NSP - cntu) + sp0, 1e-8f);
            const float dc1 = 2.0f * inter1 / fmaxf(cntu + sp1, 1e-8f);
            blob += -(dc0 + dc1) * (1.0f / 3.0f);       // dc2 == 0 exactly
        }
    }
    blob *= 1.0f / (float)(BATCH * KBLOBS);
    const float ceMean = ws[WS_CE] / (float)((long)BATCH * NSP);
    // ALPHA*CE_W = 1.0 ; BETA*DICE_W = 0.5
    out[0] = 1.0f * ceMean + 0.5f * blob;
}

extern "C" void kernel_launch(void* const* d_in, const int* in_sizes, int n_in,
                              void* d_out, int out_size, void* d_ws, size_t ws_size,
                              hipStream_t stream) {
    const float* x  = (const float*)d_in[0];
    // d_in[1] = y (binary label) is implied by ml>0; unused.
    const int*   ml = (const int*)d_in[2];
    float* ws  = (float*)d_ws;
    float* out = (float*)d_out;

    zero_ws_kernel<<<1, 64, 0, stream>>>(ws);
    blob_accum_kernel<<<NBLOCKS, 256, 0, stream>>>(x, ml, ws);
    finalize_kernel<<<1, 32, 0, stream>>>(ws, out);
}
